// MultiheadAttention_65146063946588
// MI455X (gfx1250) — matrix operand order repaired and checked by reading, not compiled, not run on hardware
//
#include <hip/hip_runtime.h>

#define L_SEQ 1024
#define BATCH 16
#define EMB   512
#define NH    8
#define HD    64
#define NUM_REL 2047

typedef __bf16 bf16;
typedef __attribute__((ext_vector_type(16))) __bf16 v16bf;
typedef __attribute__((ext_vector_type(8)))  float  v8f;

// ---------------------------------------------------------------------------
// WMMA fragment loaders (CDNA5 16-bit layouts, wave32)
// A-matrix 16x32: lane = m + 16*h ; element pair p -> K = 2*(p&3) + 8*h + (p>=4 ? 16 : 0)
// B-matrix 32x16: lane = n + 16*h ; element pair p -> K = 2*(p&3) + 8*h + (p>=4 ? 16 : 0)
// C/D 16x16 f32 : lane holds col n = lane&15 ; vgpr r holds row m = r + 8*(lane>>4)
// Tiles are stored row-major ([m][k] for A, [n][k] for B) so pairs are contiguous.
// ---------------------------------------------------------------------------
__device__ inline v16bf frag_load_A(const bf16* base, int ld, int lane) {
  const int m = lane & 15;
  const int h = lane >> 4;
  const bf16* row = base + m * ld;
  v16bf f;
#pragma unroll
  for (int p = 0; p < 8; ++p) {
    const int k = 2 * (p & 3) + 8 * h + ((p >= 4) ? 16 : 0);
    f[2 * p]     = row[k];
    f[2 * p + 1] = row[k + 1];
  }
  return f;
}

__device__ inline v16bf frag_load_B(const bf16* base, int ld, int lane) {
  const int n = lane & 15;
  const int h = lane >> 4;
  const bf16* row = base + n * ld;
  v16bf f;
#pragma unroll
  for (int p = 0; p < 8; ++p) {
    const int k = 2 * (p & 3) + 8 * h + ((p >= 4) ? 16 : 0);
    f[2 * p]     = row[k];
    f[2 * p + 1] = row[k + 1];
  }
  return f;
}

__device__ inline v8f wmma_bf16(v16bf a, v16bf b, v8f c) {
  return __builtin_amdgcn_wmma_f32_16x16x32_bf16(false, a, false, b, (short)0, c,
                                                 false, false);
}

// ---------------------------------------------------------------------------
// Kernel 1: expand RPE bias table: bias[h][l][s] = table[h, idx[l,s]]  (bf16)
// ---------------------------------------------------------------------------
__global__ __launch_bounds__(256)
void rpe_bias_kernel(const float* __restrict__ table,
                     const int* __restrict__ idx,
                     bf16* __restrict__ bias) {
  const int t   = blockIdx.x * 256 + threadIdx.x;   // over H*L*L = 8388608
  const int h   = t >> 20;                          // L*L = 2^20
  const int rem = t & 0xFFFFF;                      // l*L + s
  bias[t] = (bf16)table[h * NUM_REL + idx[rem]];
}

// ---------------------------------------------------------------------------
// Kernel 2: fused QKV projection GEMM.
//   C[m,f] = sum_e X[m,e] * Win[f,e] + bin[f],  m=(l*B+b) in 0..16383, f in 0..1535
//   Epilogue scatters into bf16 q/k/v buffers laid out [(b*H+h)*L + l]*64 + d,
//   with q scaled by 1/8.
// ---------------------------------------------------------------------------
__global__ __launch_bounds__(256)
void qkv_gemm_kernel(const float* __restrict__ q_in,
                     const float* __restrict__ k_in,
                     const float* __restrict__ v_in,
                     const float* __restrict__ w,     // (1536, 512)
                     const float* __restrict__ bvec,  // (1536)
                     bf16* __restrict__ qo,
                     bf16* __restrict__ ko,
                     bf16* __restrict__ vo) {
  __shared__ bf16 As[128][40];  // [m][k], pad 40 (even)
  __shared__ bf16 Bs[128][40];  // [n=f][k=e]

  const int tid  = threadIdx.x;
  const int lane = tid & 31;
  const int wv   = tid >> 5;     // 8 waves
  const int wm   = wv >> 2;      // 0..1 : 64-row half
  const int wn   = wv & 3;       // 0..3 : 32-col quarter
  const int bm   = blockIdx.x * 128;
  const int bn   = blockIdx.y * 128;      // f base (0..1535)
  const int which = bn >> 9;              // 0=q,1=k,2=v (128 | 512)
  const float* X = (which == 0) ? q_in : (which == 1) ? k_in : v_in;

  v8f acc[4][2] = {};

  for (int kk = 0; kk < EMB; kk += 32) {
#pragma unroll
    for (int it = 0; it < 4; ++it) {
      const int lin = tid + it * 256;     // 0..1023
      const int row = lin >> 3;           // 0..127
      const int c4  = (lin & 7) * 4;      // 0..28
      const float4 da = *(const float4*)(X + (size_t)(bm + row) * EMB + kk + c4);
      As[row][c4 + 0] = (bf16)da.x; As[row][c4 + 1] = (bf16)da.y;
      As[row][c4 + 2] = (bf16)da.z; As[row][c4 + 3] = (bf16)da.w;
      const float4 db = *(const float4*)(w + (size_t)(bn + row) * EMB + kk + c4);
      Bs[row][c4 + 0] = (bf16)db.x; Bs[row][c4 + 1] = (bf16)db.y;
      Bs[row][c4 + 2] = (bf16)db.z; Bs[row][c4 + 3] = (bf16)db.w;
    }
    __syncthreads();

    v16bf af[4], bfr[2];
#pragma unroll
    for (int i = 0; i < 4; ++i)
      af[i] = frag_load_A(&As[wm * 64 + i * 16][0], 40, lane);
#pragma unroll
    for (int j = 0; j < 2; ++j)
      bfr[j] = frag_load_B(&Bs[wn * 32 + j * 16][0], 40, lane);
#pragma unroll
    for (int i = 0; i < 4; ++i)
#pragma unroll
      for (int j = 0; j < 2; ++j)
        acc[i][j] = wmma_bf16(af[i], bfr[j], acc[i][j]);
    __syncthreads();
  }

  // Epilogue: bias add, q-scale, scatter to (b,h,l,d) bf16 buffers.
  const int hh = lane >> 4;
  const int nl = lane & 15;
  bf16* dst = (which == 0) ? qo : (which == 1) ? ko : vo;
  const float qscale = (which == 0) ? 0.125f : 1.0f;
#pragma unroll
  for (int i = 0; i < 4; ++i) {
#pragma unroll
    for (int j = 0; j < 2; ++j) {
#pragma unroll
      for (int r = 0; r < 8; ++r) {
        const int row = bm + wm * 64 + i * 16 + r + 8 * hh;  // 0..16383
        const int f   = bn + wn * 32 + j * 16 + nl;          // 0..1535
        const float val = (acc[i][j][r] + bvec[f]) * qscale;
        const int e    = f & 511;
        const int hidx = e >> 6;
        const int dd   = e & 63;
        const int l    = row >> 4;   // B = 16
        const int b    = row & 15;
        dst[((((b * NH + hidx) << 10) + l) << 6) + dd] = (bf16)val;
      }
    }
  }
}

// ---------------------------------------------------------------------------
// Kernel 3: flash attention per (b, h, 128-query block).
//   8 waves x 16 rows. Streams K/V in 32-key blocks, online softmax,
//   O += P*V via WMMA (P re-staged through per-wave LDS to become an A-frag).
//   Writes fp32 attn_out in GEMM-A layout: [(l*B + b)][h*64 + d].
// ---------------------------------------------------------------------------
__global__ __launch_bounds__(256)
void attention_kernel(const bf16* __restrict__ qb,
                      const bf16* __restrict__ kb,
                      const bf16* __restrict__ vb,
                      const bf16* __restrict__ biasb,
                      float* __restrict__ attn_out) {
  __shared__ bf16 Kt[32][72];      // [s][d]   (B-frag rows = keys)
  __shared__ bf16 Vt[64][34];      // [d][s]   (B-frag rows = d, K = keys)
  __shared__ bf16 Pb[8][16][36];   // per-wave P tile [m][s]

  const int tid  = threadIdx.x;
  const int lane = tid & 31;
  const int wv   = tid >> 5;
  const int qblk = blockIdx.x;     // 0..7
  const int h    = blockIdx.y;     // 0..7
  const int b    = blockIdx.z;     // 0..15
  const int bh   = b * NH + h;
  const bf16* qp = qb + (size_t)bh * L_SEQ * HD;
  const bf16* kp = kb + (size_t)bh * L_SEQ * HD;
  const bf16* vp = vb + (size_t)bh * L_SEQ * HD;
  const bf16* bp = biasb + (size_t)h * L_SEQ * L_SEQ;
  const int qrow0 = qblk * 128 + wv * 16;

  // Q fragments (A operand), d-chunks [0,32) and [32,64) — loaded once.
  const v16bf aq0 = frag_load_A(qp + qrow0 * HD, HD, lane);
  const v16bf aq1 = frag_load_A(qp + qrow0 * HD + 32, HD, lane);

  v8f o[4] = {};
  float mrun[8], lrun[8];
#pragma unroll
  for (int r = 0; r < 8; ++r) { mrun[r] = -1e30f; lrun[r] = 0.0f; }

  const int hh = lane >> 4;
  const int nl = lane & 15;

  for (int kb0 = 0; kb0 < L_SEQ; kb0 += 32) {
    // Stage K tile [32][64] and V^T tile [64][32]; 8 bf16 (16B) per thread.
    {
      const int row = tid >> 3;          // 0..31 (key index in block)
      const int col = (tid & 7) * 8;     // 0..56 (d)
      const uint4 dk = *(const uint4*)(kp + (kb0 + row) * HD + col);
      *(uint4*)&Kt[row][col] = dk;
      const uint4 dv = *(const uint4*)(vp + (kb0 + row) * HD + col);
      const bf16* pv = (const bf16*)&dv;
#pragma unroll
      for (int q2 = 0; q2 < 8; ++q2) Vt[col + q2][row] = pv[q2];
    }
    __syncthreads();

    // Scores S = Q * K^T for 16x32 (two 16x16 N-subtiles).
    v8f s0 = {}, s1 = {};
    {
      const v16bf b00 = frag_load_B(&Kt[0][0],   72, lane);
      const v16bf b01 = frag_load_B(&Kt[0][32],  72, lane);
      const v16bf b10 = frag_load_B(&Kt[16][0],  72, lane);
      const v16bf b11 = frag_load_B(&Kt[16][32], 72, lane);
      s0 = wmma_bf16(aq0, b00, s0);
      s0 = wmma_bf16(aq1, b01, s0);
      s1 = wmma_bf16(aq0, b10, s1);
      s1 = wmma_bf16(aq1, b11, s1);
    }
    // Relative-position bias add.
#pragma unroll
    for (int r = 0; r < 8; ++r) {
      const int lg = qrow0 + r + 8 * hh;
      s0[r] += (float)bp[lg * L_SEQ + kb0 + nl];
      s1[r] += (float)bp[lg * L_SEQ + kb0 + 16 + nl];
    }
    // Online softmax (per-row reductions across 16-lane groups).
    float pscale[8];
#pragma unroll
    for (int r = 0; r < 8; ++r) {
      float mx = fmaxf(s0[r], s1[r]);
#pragma unroll
      for (int msk = 1; msk < 16; msk <<= 1)
        mx = fmaxf(mx, __shfl_xor(mx, msk, 32));
      const float mn = fmaxf(mrun[r], mx);
      const float p0 = __expf(s0[r] - mn);
      const float p1 = __expf(s1[r] - mn);
      s0[r] = p0; s1[r] = p1;
      float ps = p0 + p1;
#pragma unroll
      for (int msk = 1; msk < 16; msk <<= 1)
        ps += __shfl_xor(ps, msk, 32);
      const float sc = __expf(mrun[r] - mn);
      lrun[r] = lrun[r] * sc + ps;
      mrun[r] = mn;
      pscale[r] = sc;
    }
#pragma unroll
    for (int t = 0; t < 4; ++t)
#pragma unroll
      for (int r = 0; r < 8; ++r) o[t][r] *= pscale[r];

    // C-layout P -> row-major LDS -> A-fragment (per-wave region, no barrier).
#pragma unroll
    for (int r = 0; r < 8; ++r) {
      const int m = r + 8 * hh;
      Pb[wv][m][nl]      = (bf16)s0[r];
      Pb[wv][m][16 + nl] = (bf16)s1[r];
    }
    const v16bf ap = frag_load_A(&Pb[wv][0][0], 36, lane);
#pragma unroll
    for (int t = 0; t < 4; ++t) {
      const v16bf bv = frag_load_B(&Vt[t * 16][0], 34, lane);
      o[t] = wmma_bf16(ap, bv, o[t]);
    }
    __syncthreads();   // before next iteration overwrites Kt/Vt
  }

  // Normalize and store fp32 in GEMM-A layout: row = l*B + b, col = h*64 + d.
#pragma unroll
  for (int r = 0; r < 8; ++r) {
    const int lg = qrow0 + r + 8 * hh;
    const float inv = 1.0f / lrun[r];
#pragma unroll
    for (int t = 0; t < 4; ++t) {
      const int dd = t * 16 + nl;
      attn_out[(size_t)(lg * BATCH + b) * EMB + h * HD + dd] = o[t][r] * inv;
    }
  }
}

// ---------------------------------------------------------------------------
// Kernel 4: output projection GEMM.
//   out[m,f] = sum_e A[m,e] * Wout[f,e] + bout[f]  (fp32 out, M=16384, N=512)
// ---------------------------------------------------------------------------
__global__ __launch_bounds__(256)
void outproj_gemm_kernel(const float* __restrict__ A,   // (16384, 512)
                         const float* __restrict__ w,   // (512, 512)
                         const float* __restrict__ bvec,
                         float* __restrict__ out) {
  __shared__ bf16 As[128][40];
  __shared__ bf16 Bs[128][40];

  const int tid  = threadIdx.x;
  const int lane = tid & 31;
  const int wv   = tid >> 5;
  const int wm   = wv >> 2;
  const int wn   = wv & 3;
  const int bm   = blockIdx.x * 128;
  const int bn   = blockIdx.y * 128;

  v8f acc[4][2] = {};

  for (int kk = 0; kk < EMB; kk += 32) {
#pragma unroll
    for (int it = 0; it < 4; ++it) {
      const int lin = tid + it * 256;
      const int row = lin >> 3;
      const int c4  = (lin & 7) * 4;
      const float4 da = *(const float4*)(A + (size_t)(bm + row) * EMB + kk + c4);
      As[row][c4 + 0] = (bf16)da.x; As[row][c4 + 1] = (bf16)da.y;
      As[row][c4 + 2] = (bf16)da.z; As[row][c4 + 3] = (bf16)da.w;
      const float4 db = *(const float4*)(w + (size_t)(bn + row) * EMB + kk + c4);
      Bs[row][c4 + 0] = (bf16)db.x; Bs[row][c4 + 1] = (bf16)db.y;
      Bs[row][c4 + 2] = (bf16)db.z; Bs[row][c4 + 3] = (bf16)db.w;
    }
    __syncthreads();

    v16bf af[4], bfr[2];
#pragma unroll
    for (int i = 0; i < 4; ++i)
      af[i] = frag_load_A(&As[wm * 64 + i * 16][0], 40, lane);
#pragma unroll
    for (int j = 0; j < 2; ++j)
      bfr[j] = frag_load_B(&Bs[wn * 32 + j * 16][0], 40, lane);
#pragma unroll
    for (int i = 0; i < 4; ++i)
#pragma unroll
      for (int j = 0; j < 2; ++j)
        acc[i][j] = wmma_bf16(af[i], bfr[j], acc[i][j]);
    __syncthreads();
  }

  const int hh = lane >> 4;
  const int nl = lane & 15;
#pragma unroll
  for (int i = 0; i < 4; ++i) {
#pragma unroll
    for (int j = 0; j < 2; ++j) {
#pragma unroll
      for (int r = 0; r < 8; ++r) {
        const int row = bm + wm * 64 + i * 16 + r + 8 * hh;
        const int f   = bn + wn * 32 + j * 16 + nl;
        out[(size_t)row * EMB + f] = acc[i][j][r] + bvec[f];
      }
    }
  }
}

// ---------------------------------------------------------------------------
// Launcher
// ---------------------------------------------------------------------------
extern "C" void kernel_launch(void* const* d_in, const int* in_sizes, int n_in,
                              void* d_out, int out_size, void* d_ws, size_t ws_size,
                              hipStream_t stream) {
  (void)in_sizes; (void)n_in; (void)out_size; (void)ws_size;
  const float* query = (const float*)d_in[0];
  const float* key   = (const float*)d_in[1];
  const float* value = (const float*)d_in[2];
  const float* w_in  = (const float*)d_in[3];
  const float* b_in  = (const float*)d_in[4];
  const float* w_out = (const float*)d_in[5];
  const float* b_out = (const float*)d_in[6];
  const float* rpe_t = (const float*)d_in[7];
  const int*   rpe_i = (const int*)d_in[8];
  float* out = (float*)d_out;

  char* ws = (char*)d_ws;
  bf16*  qb       = (bf16*)(ws);                     // 16 MB
  bf16*  kb       = (bf16*)(ws + ((size_t)16 << 20)); // 16 MB
  bf16*  vb       = (bf16*)(ws + ((size_t)32 << 20)); // 16 MB
  bf16*  biasb    = (bf16*)(ws + ((size_t)48 << 20)); // 16 MB
  float* attn_out = (float*)(ws + ((size_t)64 << 20)); // 32 MB

  rpe_bias_kernel<<<dim3((NH * L_SEQ * L_SEQ) / 256), 256, 0, stream>>>(
      rpe_t, rpe_i, biasb);
  qkv_gemm_kernel<<<dim3(128, 12), 256, 0, stream>>>(
      query, key, value, w_in, b_in, qb, kb, vb);
  attention_kernel<<<dim3(8, NH, BATCH), 256, 0, stream>>>(
      qb, kb, vb, biasb, attn_out);
  outproj_gemm_kernel<<<dim3(128, 4), 256, 0, stream>>>(
      attn_out, w_out, b_out, out);
}
